// DynaResidualBlock_14903536517595
// MI455X (gfx1250) — matrix operand order repaired and checked by reading, not compiled, not run on hardware
//
#include <hip/hip_runtime.h>

// ---------------------------------------------------------------------------
// DynaResidualBlock for MI455X (gfx1250, wave32, WMMA bf16).
//  - Hypernet: 6x M=16 GEMMs via v_wmma_f32_16x16x32_bf16 (bandwidth bound),
//    templated on (K,N) so strided weight loads fold into immediate offsets.
//    Final GEMM emits ksv as packed bf16 (halves the dyna-side L2 stream).
//  - Dyna block: fully fused 4-stage per-sample 1x1-conv chain; bf16 kernel
//    blob staged via global_load_async_to_lds_b128 (ASYNCcnt), x read once
//    (non-temporal), intermediates never hit HBM.
//  - All LDS fragment traffic is b128; f32->bf16 packs are single
//    v_cvt_pk_bf16_f32 ops via inline asm.
// ---------------------------------------------------------------------------

typedef __attribute__((ext_vector_type(16))) __bf16        bf16x16;
typedef __attribute__((ext_vector_type(8)))  float         v8f;
typedef __attribute__((ext_vector_type(8)))  unsigned int  v8u;
typedef __attribute__((ext_vector_type(4)))  unsigned int  v4u;
typedef __attribute__((ext_vector_type(2)))  unsigned int  v2u;

#define B_    16
#define FIN_  64
#define FOUT_ 64
#define FHID_ 128
#define HW_   16384
#define KTOT_ 37248

// one packed f32x2 -> bf16x2 convert (native gfx1250 instruction)
__device__ __forceinline__ unsigned pk2(float lo, float hi) {
  unsigned r;
  asm("v_cvt_pk_bf16_f32 %0, %1, %2" : "=v"(r) : "v"(lo), "v"(hi));
  return r;
}
__device__ __forceinline__ unsigned short f2bf(float f) {
  return __builtin_bit_cast(unsigned short, (__bf16)f);
}
__device__ __forceinline__ float bf2f(unsigned short u) {
  return __builtin_bit_cast(float, (unsigned)u << 16);   // exact
}

// CDNA5 async LDS copy: LDS[lds_byte_off] = MEM[g .. g+15], ASYNCcnt-tracked
__device__ __forceinline__ void async_ld_b128(unsigned lds_byte_off, const void* g) {
  asm volatile("global_load_async_to_lds_b128 %0, %1, off"
               :: "v"(lds_byte_off), "v"(g) : "memory");
}
__device__ __forceinline__ void wait_async0() {
  asm volatile("s_wait_asynccnt 0x0" ::: "memory");
}

// A-fragment (16x32 bf16). lanes 0-15 / 16-31 both hold M=lane&15;
// VGPR v holds K pair {base,base+1}, base = half*16 + 2*v.
// K run {kk+half*16 .. +15} is 8 contiguous dwords in row-major LDS -> 2x b128.
__device__ __forceinline__ bf16x16 load_a_lds(const unsigned short* base_, int m,
                                              int ld, int kk, int half) {
  const unsigned short* row = base_ + m * ld + kk + half * 16;
  v4u a0 = *(const v4u*)(row);
  v4u a1 = *(const v4u*)(row + 8);
  v8u p;
  p[0] = a0[0]; p[1] = a0[1]; p[2] = a0[2]; p[3] = a0[3];
  p[4] = a1[0]; p[5] = a1[1]; p[6] = a1[2]; p[7] = a1[3];
  return __builtin_bit_cast(bf16x16, p);
}

// B-fragment (32x16 bf16) from LDS activations stored [col][channel]:
// K run {kk+half*16 .. +15} is 8 contiguous dwords -> 2x b128.
__device__ __forceinline__ bf16x16 load_b_lds(const unsigned short* base_, int n,
                                              int ld, int kk, int half) {
  const unsigned short* col = base_ + n * ld + kk + half * 16;
  v4u b0 = *(const v4u*)(col);
  v4u b1 = *(const v4u*)(col + 8);
  v8u p;
  p[0] = b0[0]; p[1] = b0[1]; p[2] = b0[2]; p[3] = b0[3];
  p[4] = b1[0]; p[5] = b1[1]; p[6] = b1[2]; p[7] = b1[3];
  return __builtin_bit_cast(bf16x16, p);
}

__device__ __forceinline__ v8f wmma_bf16(bf16x16 a, bf16x16 b, v8f c) {
  return __builtin_amdgcn_wmma_f32_16x16x32_bf16(
      /*neg_a=*/false, a, /*neg_b=*/false, b,
      /*c_mod=*/(short)0, c, /*reuse_a=*/false, /*reuse_b=*/false);
}

// ---------------------------------------------------------------------------
// Generic hypernet GEMM: out[16,N] = op(A[16,K] @ W[K,N] + bias + add)
// grid.x = N/128, 256 threads = 8 waves, one 16x16 C tile per wave.
// K,N compile-time: weight-load strides become immediate global_load offsets.
// OUTBF16: store packed bf16 (used for the final ksv producer).
// ---------------------------------------------------------------------------
template <int K, int N, bool OUTBF16>
__global__ __launch_bounds__(256)
void gemm_m16_kernel(const float* __restrict__ A, const float* __restrict__ W,
                     const float* __restrict__ bias, const float* __restrict__ add,
                     void* __restrict__ outv, int relu) {
  __shared__ unsigned short aLds[16 * K];

  int tid = threadIdx.x;
#pragma unroll 4
  for (int i = tid; i < (16 * K) >> 2; i += 256) {   // float4 stream -> packed bf16
    float4 f = ((const float4*)A)[i];
    v2u w; w[0] = pk2(f.x, f.y); w[1] = pk2(f.z, f.w);
    *(v2u*)(aLds + i * 4) = w;
  }
  __syncthreads();

  int wave = tid >> 5, lane = tid & 31, half = lane >> 4, col = lane & 15;
  int n = blockIdx.x * 128 + wave * 16 + col;

  const float* wp = W + (size_t)(half * 16) * N + n;  // lane's weight column base
  v8f acc = {};
#pragma unroll 2
  for (int kk = 0; kk < K; kk += 32) {
    bf16x16 a = load_a_lds(aLds, col, K, kk, half);
    v8u p;
#pragma unroll
    for (int v = 0; v < 8; ++v)                       // const offsets -> imm IOFFSET
      p[v] = pk2(wp[2 * v * N], wp[(2 * v + 1) * N]);
    wp += (size_t)32 * N;
    acc = wmma_bf16(a, __builtin_bit_cast(bf16x16, p), acc);
  }

#pragma unroll
  for (int r = 0; r < 8; ++r) {
    int m = r + 8 * half;                 // C layout: M = vgpr + 8*(lane>=16)
    float v = acc[r];
    if (bias) v += bias[n];
    if (add)  v += add[(size_t)m * N + n];
    if (relu) v = v > 0.f ? v : 0.f;
    if (OUTBF16) ((unsigned short*)outv)[(size_t)m * N + n] = f2bf(v);
    else         ((float*)outv)[(size_t)m * N + n] = v;
  }
}

// ---------------------------------------------------------------------------
// Fused dyna block. grid = (HW/128, B), 256 threads = 8 waves.
// Each wave owns 16 spatial columns and runs XS -> H1 -> H2 -> OUT privately;
// bf16 per-sample kernels async-copied to LDS; x tile converted in-register.
// Dynamic LDS = 124416 B.
// ---------------------------------------------------------------------------
__global__ __launch_bounds__(256)
void dyna_fused_kernel(const float* __restrict__ x,
                       const unsigned short* __restrict__ ksv16,
                       float* __restrict__ out) {
  extern __shared__ unsigned short smem[];
  unsigned short* kin  = smem;             // 128*64   = 8192
  unsigned short* kmid = smem + 8192;      // 128*128  = 16384
  unsigned short* kout = smem + 24576;     // 64*128   = 8192
  unsigned short* ksh  = smem + 32768;     // 64*64    = 4096
  unsigned short* xt   = smem + 36864;     // [col 0..127][ch 0..63]
  unsigned short* hbuf = smem + 45056;     // [col 0..127][ch 0..127]
  float* bias_f = (float*)(smem + 61440);  // 384 floats
  float* b_in  = bias_f;
  float* b_mid = bias_f + 128;
  float* b_out = bias_f + 256;
  float* b_sh  = bias_f + 320;

  int b   = blockIdx.y;
  int s0  = blockIdx.x * 128;
  int tid = threadIdx.x;
  const unsigned short* kb = ksv16 + (size_t)b * KTOT_;

  // async-DMA the 72KB bf16 kernel blob (k_in|k_mid|k_out|k_sh) into LDS
  unsigned lds0 = (unsigned)(uintptr_t)smem;
#pragma unroll 2
  for (int i = tid; i < 36864 / 8; i += 256)          // 4608x b128
    async_ld_b128(lds0 + i * 16, kb + i * 8);
  // biases arrive bf16; expand exactly (<<16) to f32 in LDS
  for (int i = tid; i < 384; i += 256) bias_f[i] = bf2f(kb[36864 + i]);
  // x tile (read-once -> non-temporal), transposed to [col][channel]
#pragma unroll 4
  for (int i = tid; i < (FIN_ / 2) * 128; i += 256) {
    int c2 = (i >> 7) * 2, j = i & 127;
    const float* xb = x + (size_t)b * FIN_ * HW_ + s0 + j;
    float f0 = __builtin_nontemporal_load(xb + (size_t)c2 * HW_);
    float f1 = __builtin_nontemporal_load(xb + (size_t)(c2 + 1) * HW_);
    *(unsigned*)(xt + j * FIN_ + c2) = pk2(f0, f1);
  }
  wait_async0();
  __syncthreads();

  int wave = tid >> 5, lane = tid & 31, half = lane >> 4, col = lane & 15;
  int n = wave * 16 + col;                 // this lane's spatial column

  // --- shortcut XS = k_sh @ X  (M=64, K=64) ---------------------------------
  v8f accS[4] = {};
#pragma unroll
  for (int kk = 0; kk < FIN_; kk += 32) {
    bf16x16 bx = load_b_lds(xt, n, FIN_, kk, half);
#pragma unroll
    for (int t = 0; t < 4; ++t)
      accS[t] = wmma_bf16(load_a_lds(ksh + t * 16 * FIN_, col, FIN_, kk, half), bx, accS[t]);
  }

  // --- H1 = relu(k_in @ X + b_in)  (M=128, K=64) ----------------------------
  {
    v8f acc1[8] = {};
#pragma unroll
    for (int kk = 0; kk < FIN_; kk += 32) {
      bf16x16 bx = load_b_lds(xt, n, FIN_, kk, half);
#pragma unroll
      for (int t = 0; t < 8; ++t)
        acc1[t] = wmma_bf16(load_a_lds(kin + t * 16 * FIN_, col, FIN_, kk, half), bx, acc1[t]);
    }
#pragma unroll
    for (int t = 0; t < 8; ++t) {         // 8 consecutive channels -> 1x b128
      int ch0 = t * 16 + 8 * half;
      v4u w;
#pragma unroll
      for (int rp = 0; rp < 4; ++rp) {
        float v0 = acc1[t][2 * rp]     + b_in[ch0 + 2 * rp];
        float v1 = acc1[t][2 * rp + 1] + b_in[ch0 + 2 * rp + 1];
        w[rp] = pk2(v0 > 0.f ? v0 : 0.f, v1 > 0.f ? v1 : 0.f);
      }
      *(v4u*)(hbuf + n * FHID_ + ch0) = w;
    }
  }
  __syncthreads();

  // --- H2 = relu(k_mid @ H1 + b_mid)  (M=128, K=128) ------------------------
  {
    v8f acc2[8] = {};
#pragma unroll
    for (int kk = 0; kk < FHID_; kk += 32) {
      bf16x16 bx = load_b_lds(hbuf, n, FHID_, kk, half);
#pragma unroll
      for (int t = 0; t < 8; ++t)
        acc2[t] = wmma_bf16(load_a_lds(kmid + t * 16 * FHID_, col, FHID_, kk, half), bx, acc2[t]);
    }
    __syncthreads();   // all H1 reads done before hbuf is overwritten
#pragma unroll
    for (int t = 0; t < 8; ++t) {
      int ch0 = t * 16 + 8 * half;
      v4u w;
#pragma unroll
      for (int rp = 0; rp < 4; ++rp) {
        float v0 = acc2[t][2 * rp]     + b_mid[ch0 + 2 * rp];
        float v1 = acc2[t][2 * rp + 1] + b_mid[ch0 + 2 * rp + 1];
        w[rp] = pk2(v0 > 0.f ? v0 : 0.f, v1 > 0.f ? v1 : 0.f);
      }
      *(v4u*)(hbuf + n * FHID_ + ch0) = w;
    }
  }
  __syncthreads();

  // --- OUT = k_out @ H2 + b_out + XS + b_sh  (M=64, K=128) ------------------
  v8f accO[4] = {};
#pragma unroll
  for (int kk = 0; kk < FHID_; kk += 32) {
    bf16x16 bx = load_b_lds(hbuf, n, FHID_, kk, half);
#pragma unroll
    for (int t = 0; t < 4; ++t)
      accO[t] = wmma_bf16(load_a_lds(kout + t * 16 * FHID_, col, FHID_, kk, half), bx, accO[t]);
  }
#pragma unroll
  for (int t = 0; t < 4; ++t)
#pragma unroll
    for (int r = 0; r < 8; ++r) {
      int ch = t * 16 + r + 8 * half;
      float v = accO[t][r] + accS[t][r] + b_out[ch] + b_sh[ch];
      __builtin_nontemporal_store(v, out + ((size_t)b * FOUT_ + ch) * HW_ + s0 + n);
    }
}

// ---------------------------------------------------------------------------
extern "C" void kernel_launch(void* const* d_in, const int* in_sizes, int n_in,
                              void* d_out, int out_size, void* d_ws, size_t ws_size,
                              hipStream_t stream) {
  (void)in_sizes; (void)n_in; (void)out_size; (void)ws_size;

  const float* x     = (const float*)d_in[0];
  const float* lat   = (const float*)d_in[1];
  const float* dk_w  = (const float*)d_in[2];
  const float* dk_b  = (const float*)d_in[3];
  const float* l1_w1 = (const float*)d_in[4];
  const float* l1_b1 = (const float*)d_in[5];
  const float* l1_w2 = (const float*)d_in[6];
  const float* l1_b2 = (const float*)d_in[7];
  const float* l2_w1 = (const float*)d_in[8];
  const float* l2_b1 = (const float*)d_in[9];
  const float* l2_w2 = (const float*)d_in[10];
  const float* l2_b2 = (const float*)d_in[11];
  const float* l2_ws = (const float*)d_in[12];
  float* out = (float*)d_out;

  float* ws    = (float*)d_ws;
  float* h1    = ws;               // 16*512
  float* t1    = h1 + 16 * 512;    // 16*1024
  float* h2    = t1 + 16 * 1024;   // 16*512
  float* rr    = h2 + 16 * 512;    // 16*1024
  float* ksv   = rr + 16 * 1024;   // 16*37248 f32 (partial sum)
  unsigned short* ksv16 = (unsigned short*)(ksv + (size_t)16 * KTOT_); // 16*37248 bf16

  // hypernet: h1 = lat@dk_w + dk_b
  gemm_m16_kernel<512, 512, false><<<4, 256, 0, stream>>>(lat, dk_w, dk_b, nullptr, h1, 0);
  // LRB1: h2 = h1 + relu(h1@w1+b1)@w2 + b2
  gemm_m16_kernel<512, 1024, false><<<8, 256, 0, stream>>>(h1, l1_w1, l1_b1, nullptr, t1, 1);
  gemm_m16_kernel<1024, 512, false><<<4, 256, 0, stream>>>(t1, l1_w2, l1_b2, h1, h2, 0);
  // LRB2: ksv = h2@l2_ws + relu(h2@w1+b1)@w2 + b2   (fat, L2/HBM bound)
  gemm_m16_kernel<512, 1024, false><<<8, 256, 0, stream>>>(h2, l2_w1, l2_b1, nullptr, rr, 1);
  gemm_m16_kernel<512, KTOT_, false><<<291, 256, 0, stream>>>(h2, l2_ws, l2_b2, nullptr, ksv, 0);
  gemm_m16_kernel<1024, KTOT_, true><<<291, 256, 0, stream>>>(rr, l2_w2, nullptr, ksv, ksv16, 0);

  // fused dyna block
  const int smem_bytes = 61440 * 2 + 384 * 4;  // 124416 B (<320 KB/WGP)
  hipFuncSetAttribute((const void*)dyna_fused_kernel,
                      hipFuncAttributeMaxDynamicSharedMemorySize, smem_bytes);
  dyna_fused_kernel<<<dim3(HW_ / 128, B_), 256, smem_bytes, stream>>>(x, ksv16, out);
}
